// AttentionHead_46531675685093
// MI455X (gfx1250) — matrix operand order repaired and checked by reading, not compiled, not run on hardware
//
#include <hip/hip_runtime.h>

typedef __attribute__((ext_vector_type(16))) _Float16 v16h;
typedef __attribute__((ext_vector_type(8)))  _Float16 v8h;
typedef __attribute__((ext_vector_type(4)))  _Float16 v4h;
typedef __attribute__((ext_vector_type(8)))  float    v8f;
typedef __attribute__((ext_vector_type(4)))  int      v4i;

#define NB   4
#define SEQ  4096
#define DIM  1024
#define DKV  64

#if defined(__has_builtin)
#if __has_builtin(__builtin_amdgcn_global_load_async_to_lds_b128) && \
    __has_builtin(__builtin_amdgcn_s_wait_asynccnt)
#define USE_ASYNC_LDS 1
#endif
#endif

static __device__ __forceinline__ v8f wmma16(v16h a, v16h b, v8f c) {
  // D = A(16x32 f16) * B(32x16 f16) + C(16x16 f32)
  return __builtin_amdgcn_wmma_f32_16x16x32_f16(false, a, false, b, (short)0, c,
                                                false, false);
}

// ---------------------------------------------------------------------------
// Kernel 0: convert the three weight matrices to f16 once (384 KB, L2-resident)
// Wh layout: [3][64][1024] halves, order q,k,v.
// ---------------------------------------------------------------------------
__global__ __launch_bounds__(256) void cvt_w_kernel(
    const float* __restrict__ Wq, const float* __restrict__ Wk,
    const float* __restrict__ Wv, _Float16* __restrict__ Wh)
{
  const int i = (blockIdx.x * 256 + threadIdx.x) * 4;   // over 64*1024 floats
  if (i >= 64 * DIM) return;
  float4 aq = *(const float4*)(Wq + i);
  float4 ak = *(const float4*)(Wk + i);
  float4 av = *(const float4*)(Wv + i);
  v4h hq, hk, hv;
  hq[0]=(_Float16)aq.x; hq[1]=(_Float16)aq.y; hq[2]=(_Float16)aq.z; hq[3]=(_Float16)aq.w;
  hk[0]=(_Float16)ak.x; hk[1]=(_Float16)ak.y; hk[2]=(_Float16)ak.z; hk[3]=(_Float16)ak.w;
  hv[0]=(_Float16)av.x; hv[1]=(_Float16)av.y; hv[2]=(_Float16)av.z; hv[3]=(_Float16)av.w;
  *(v4h*)(Wh + i)              = hq;
  *(v4h*)(Wh + 64 * DIM + i)   = hk;
  *(v4h*)(Wh + 128 * DIM + i)  = hv;
}

// ---------------------------------------------------------------------------
// Kernel 1: fused QKV projection.  y = x @ W^T + b  (x f32 -> QKV f16)
// 8 waves x 16 rows = 128 rows/block.  f16 weights staged in LDS per 128-wide
// k-chunk; inner loop is 4 k-steps x 12 WMMA with only x coming from global.
// ---------------------------------------------------------------------------
__global__ __launch_bounds__(256) void qkv_proj_kernel(
    const float* __restrict__ x,
    const _Float16* __restrict__ Wh,
    const float* __restrict__ bq, const float* __restrict__ bk,
    const float* __restrict__ bv,
    _Float16* __restrict__ Qh, _Float16* __restrict__ Kh,
    _Float16* __restrict__ Vh)
{
  __shared__ __align__(16) _Float16 WL[3 * 64 * 128];   // 48 KB

  const int lane = threadIdx.x & 31;
  const int wave = threadIdx.x >> 5;
  const int nn   = lane & 15;
  const int g    = lane >> 4;
  const int r0   = blockIdx.x * 128 + wave * 16;

  v8f zero;
  #pragma unroll
  for (int j = 0; j < 8; ++j) zero[j] = 0.0f;
  v8f accQ[4], accK[4], accV[4];
  #pragma unroll
  for (int t = 0; t < 4; ++t) { accQ[t] = zero; accK[t] = zero; accV[t] = zero; }

  float bqv[4], bkv[4], bvv[4];
  #pragma unroll
  for (int t = 0; t < 4; ++t) {
    bqv[t] = bq[16 * t + nn];
    bkv[t] = bk[16 * t + nn];
    bvv[t] = bv[16 * t + nn];
  }

  const float* xrow = x + (size_t)(r0 + nn) * DIM;

  // staging roles: thread t copies 32 halves per matrix per chunk
  const int srow = threadIdx.x >> 2;          // 0..63
  const int scol = (threadIdx.x & 3) * 32;    // 0,32,64,96

  for (int kc = 0; kc < DIM; kc += 128) {
    __syncthreads();                          // previous chunk fully consumed
    #pragma unroll
    for (int m = 0; m < 3; ++m) {
      const _Float16* src = Wh + ((size_t)m * 64 + srow) * DIM + kc + scol;
      _Float16* dst = &WL[m * 64 * 128 + srow * 128 + scol];
      *(v8h*)(dst)      = *(const v8h*)(src);
      *(v8h*)(dst + 8)  = *(const v8h*)(src + 8);
      *(v8h*)(dst + 16) = *(const v8h*)(src + 16);
      *(v8h*)(dst + 24) = *(const v8h*)(src + 24);
    }
    __syncthreads();

    #pragma unroll
    for (int ki = 0; ki < 128; ki += 32) {
      // ---- A fragment from x (f32 stream -> f16) ----
      const float* xp = xrow + kc + ki + 8 * g;
      float4 c0 = *(const float4*)(xp);
      float4 c1 = *(const float4*)(xp + 4);
      float4 c2 = *(const float4*)(xp + 16);
      float4 c3 = *(const float4*)(xp + 20);
      v16h A;
      A[0]=(_Float16)c0.x;  A[1]=(_Float16)c0.y;  A[2]=(_Float16)c0.z;  A[3]=(_Float16)c0.w;
      A[4]=(_Float16)c1.x;  A[5]=(_Float16)c1.y;  A[6]=(_Float16)c1.z;  A[7]=(_Float16)c1.w;
      A[8]=(_Float16)c2.x;  A[9]=(_Float16)c2.y;  A[10]=(_Float16)c2.z; A[11]=(_Float16)c2.w;
      A[12]=(_Float16)c3.x; A[13]=(_Float16)c3.y; A[14]=(_Float16)c3.z; A[15]=(_Float16)c3.w;

      const int kb = ki + 8 * g;   // B-frag: k = kb..kb+7 and kb+16..kb+23 in LDS row
      #pragma unroll
      for (int t = 0; t < 4; ++t) {
        const int row = 16 * t + nn;
        v16h Bq, Bk2, Bv2;
        const _Float16* p0 = &WL[row * 128 + kb];
        ((v8h*)&Bq)[0]  = *(const v8h*)(p0);
        ((v8h*)&Bq)[1]  = *(const v8h*)(p0 + 16);
        const _Float16* p1 = &WL[64 * 128 + row * 128 + kb];
        ((v8h*)&Bk2)[0] = *(const v8h*)(p1);
        ((v8h*)&Bk2)[1] = *(const v8h*)(p1 + 16);
        const _Float16* p2 = &WL[2 * 64 * 128 + row * 128 + kb];
        ((v8h*)&Bv2)[0] = *(const v8h*)(p2);
        ((v8h*)&Bv2)[1] = *(const v8h*)(p2 + 16);
        accQ[t] = wmma16(A, Bq,  accQ[t]);
        accK[t] = wmma16(A, Bk2, accK[t]);
        accV[t] = wmma16(A, Bv2, accV[t]);
      }
    }
  }

  // ---- store C/D layout: lane col = nn, VGPR j -> row j + 8g ----
  #pragma unroll
  for (int t = 0; t < 4; ++t) {
    #pragma unroll
    for (int j = 0; j < 8; ++j) {
      const size_t row = (size_t)(r0 + j + 8 * g);
      const int    col = 16 * t + nn;
      Qh[row * DKV + col] = (_Float16)(accQ[t][j] + bqv[t]);
      Kh[row * DKV + col] = (_Float16)(accK[t][j] + bkv[t]);
      Vh[row * DKV + col] = (_Float16)(accV[t][j] + bvv[t]);
    }
  }
}

// ---------------------------------------------------------------------------
// Kernel 2: flash attention.  Block = 128 query rows (8 waves x 16 rows).
// Stream 64-key tiles: K row-major in LDS (async-to-LDS when available),
// V transposed in LDS so all WMMA B-fragments are contiguous ds_load_b128
// pairs.  Online softmax in f32; P round-trips through wave-private LDS.
// ---------------------------------------------------------------------------
__global__ __launch_bounds__(256) void flash_attn_kernel(
    const _Float16* __restrict__ Qh, const _Float16* __restrict__ Kh,
    const _Float16* __restrict__ Vh, float* __restrict__ out)
{
  __shared__ __align__(16) _Float16 Kt[64 * 64];       // [key][dk]   8 KB
  __shared__ __align__(16) _Float16 Vt[64 * 64];       // [vdim][key] 8 KB
  __shared__ __align__(16) _Float16 Pb[8 * 16 * 64];   // P scratch  16 KB

  const int lane = threadIdx.x & 31;
  const int wave = threadIdx.x >> 5;
  const int nn   = lane & 15;
  const int g    = lane >> 4;

  const int b  = blockIdx.x >> 5;             // SEQ/128 = 32 blocks per batch
  const int q0 = (blockIdx.x & 31) * 128;
  const size_t bbase = (size_t)b * SEQ;

  // ---- load this wave's Q A-fragments (rows r0..r0+15, k split 0/32) ----
  const int r0 = q0 + wave * 16;
  v16h Aq[2];
  const _Float16* qrow = Qh + (bbase + r0 + nn) * DKV;
  #pragma unroll
  for (int f = 0; f < 2; ++f) {
    const _Float16* qp = qrow + 32 * f + 8 * g;
    ((v8h*)&Aq[f])[0] = *(const v8h*)(qp);
    ((v8h*)&Aq[f])[1] = *(const v8h*)(qp + 16);
  }

  v8f zero;
  #pragma unroll
  for (int j = 0; j < 8; ++j) zero[j] = 0.0f;
  v8f accO[4];
  #pragma unroll
  for (int t = 0; t < 4; ++t) accO[t] = zero;
  float rm[8], rs[8];
  #pragma unroll
  for (int j = 0; j < 8; ++j) { rm[j] = -__builtin_inff(); rs[j] = 0.0f; }

  _Float16* Pw = Pb + wave * 16 * 64;
  const float scale = 0.125f;                 // 1/sqrt(64)

  // staging roles (256 threads cover a 64x64 f16 tile, 16 halves each)
  const int ss  = threadIdx.x >> 2;           // key row 0..63
  const int dch = (threadIdx.x & 3) * 16;     // 16-wide chunk

  for (int kb = 0; kb < SEQ; kb += 64) {
    __syncthreads();                          // LDS reuse fence
    // ---- stage K (row-major) ----
    const _Float16* ksrc = Kh + (bbase + kb + ss) * DKV + dch;
#ifdef USE_ASYNC_LDS
    __builtin_amdgcn_global_load_async_to_lds_b128(
        (v4i*)ksrc, (v4i*)&Kt[ss * 64 + dch], 0, 0);
    __builtin_amdgcn_global_load_async_to_lds_b128(
        (v4i*)(ksrc + 8), (v4i*)&Kt[ss * 64 + dch + 8], 0, 0);
#else
    *(v8h*)&Kt[ss * 64 + dch]     = *(const v8h*)(ksrc);
    *(v8h*)&Kt[ss * 64 + dch + 8] = *(const v8h*)(ksrc + 8);
#endif
    // ---- stage V transposed ----
    const _Float16* vsrc = Vh + (bbase + kb + ss) * DKV + dch;
    v8h v0 = *(const v8h*)(vsrc);
    v8h v1 = *(const v8h*)(vsrc + 8);
    #pragma unroll
    for (int i = 0; i < 8; ++i) {
      Vt[(dch + i) * 64 + ss]     = v0[i];
      Vt[(dch + 8 + i) * 64 + ss] = v1[i];
    }
    // prefetch next tile while we compute on this one
    const int kn = (kb + 64 < SEQ) ? kb + 64 : kb;
    __builtin_prefetch(Kh + (bbase + kn + ss) * DKV + dch, 0, 0);
    __builtin_prefetch(Vh + (bbase + kn + ss) * DKV + dch, 0, 0);
#ifdef USE_ASYNC_LDS
    __builtin_amdgcn_s_wait_asynccnt(0);
#endif
    __syncthreads();

    // ---- scores S = Q K^T for 4 column tiles of 16 keys ----
    v8f st[4];
    #pragma unroll
    for (int t = 0; t < 4; ++t) {
      v8f sc = zero;
      #pragma unroll
      for (int f = 0; f < 2; ++f) {
        v16h Bk;
        const _Float16* kp = &Kt[(16 * t + nn) * 64 + 32 * f + 8 * g];
        ((v8h*)&Bk)[0] = *(const v8h*)(kp);
        ((v8h*)&Bk)[1] = *(const v8h*)(kp + 16);
        sc = wmma16(Aq[f], Bk, sc);
      }
      #pragma unroll
      for (int j = 0; j < 8; ++j) sc[j] *= scale;
      st[t] = sc;
    }

    // ---- online softmax (rows j+8g live across 16 lanes of this group) ----
    #pragma unroll
    for (int j = 0; j < 8; ++j) {
      float v = fmaxf(fmaxf(st[0][j], st[1][j]), fmaxf(st[2][j], st[3][j]));
      v = fmaxf(v, __shfl_xor(v, 1, 32));
      v = fmaxf(v, __shfl_xor(v, 2, 32));
      v = fmaxf(v, __shfl_xor(v, 4, 32));
      v = fmaxf(v, __shfl_xor(v, 8, 32));
      const float nm   = fmaxf(rm[j], v);
      const float corr = __expf(rm[j] - nm);
      rm[j] = nm;
      rs[j] *= corr;
      accO[0][j] *= corr; accO[1][j] *= corr;
      accO[2][j] *= corr; accO[3][j] *= corr;
    }

    float psum[8];
    #pragma unroll
    for (int j = 0; j < 8; ++j) psum[j] = 0.0f;
    #pragma unroll
    for (int t = 0; t < 4; ++t) {
      #pragma unroll
      for (int j = 0; j < 8; ++j) {
        const float p = __expf(st[t][j] - rm[j]);
        psum[j] += p;
        Pw[(j + 8 * g) * 64 + 16 * t + nn] = (_Float16)p;   // C-layout -> LDS
      }
    }
    #pragma unroll
    for (int j = 0; j < 8; ++j) {
      float v = psum[j];
      v += __shfl_xor(v, 1, 32);
      v += __shfl_xor(v, 2, 32);
      v += __shfl_xor(v, 4, 32);
      v += __shfl_xor(v, 8, 32);
      rs[j] += v;
    }

    // ---- O += P V : reload P as A-fragments (wave-private LDS, in-order) ----
    v16h Ap[2];
    #pragma unroll
    for (int f = 0; f < 2; ++f) {
      const _Float16* pp = Pw + nn * 64 + 32 * f + 8 * g;
      ((v8h*)&Ap[f])[0] = *(const v8h*)(pp);
      ((v8h*)&Ap[f])[1] = *(const v8h*)(pp + 16);
    }
    #pragma unroll
    for (int t = 0; t < 4; ++t) {
      #pragma unroll
      for (int f = 0; f < 2; ++f) {
        v16h Bv;
        const _Float16* vp = &Vt[(16 * t + nn) * 64 + 32 * f + 8 * g];
        ((v8h*)&Bv)[0] = *(const v8h*)(vp);
        ((v8h*)&Bv)[1] = *(const v8h*)(vp + 16);
        accO[t] = wmma16(Ap[f], Bv, accO[t]);
      }
    }
  }

  // ---- finalize: divide by row sums, write f32 output ----
  #pragma unroll
  for (int t = 0; t < 4; ++t) {
    #pragma unroll
    for (int j = 0; j < 8; ++j) {
      const size_t row = bbase + r0 + j + 8 * g;
      out[row * DKV + 16 * t + nn] = accO[t][j] / rs[j];
    }
  }
}

// ---------------------------------------------------------------------------
extern "C" void kernel_launch(void* const* d_in, const int* in_sizes, int n_in,
                              void* d_out, int out_size, void* d_ws, size_t ws_size,
                              hipStream_t stream) {
  const float* x  = (const float*)d_in[0];
  const float* Wq = (const float*)d_in[1];
  const float* bq = (const float*)d_in[2];
  const float* Wk = (const float*)d_in[3];
  const float* bk = (const float*)d_in[4];
  const float* Wv = (const float*)d_in[5];
  const float* bv = (const float*)d_in[6];
  float* out = (float*)d_out;

  _Float16* Qh = (_Float16*)d_ws;                       // 2 MB
  _Float16* Kh = Qh + (size_t)NB * SEQ * DKV;           // 2 MB
  _Float16* Vh = Kh + (size_t)NB * SEQ * DKV;           // 2 MB
  _Float16* Wh = Vh + (size_t)NB * SEQ * DKV;           // 384 KB

  cvt_w_kernel<<<64, 256, 0, stream>>>(Wq, Wk, Wv, Wh);

  const int blocks = NB * SEQ / 128;                    // 128 blocks
  qkv_proj_kernel<<<blocks, 256, 0, stream>>>(x, Wh, bq, bk, bv, Qh, Kh, Vh);
  flash_attn_kernel<<<blocks, 256, 0, stream>>>(Qh, Kh, Vh, out);
}
